// AggLayer_3289944949009
// MI455X (gfx1250) — matrix operand order repaired and checked by reading, not compiled, not run on hardware
//
#include <hip/hip_runtime.h>

// ---------------------------------------------------------------------------
// CDNA5 (gfx1250) implementation of the 6-layer edge-conditioned TAGConv GNN.
// wave32, WMMA f32_16x16x32_f16 for all 32-wide GEMMs, f32 atomics for the
// segment-sum hops, device-side binary-search top-k (graph-capture safe).
// Template-specialized GEMM epilogues: full-tile waves run branch-free.
// ---------------------------------------------------------------------------

typedef __attribute__((ext_vector_type(16))) _Float16 v16h;
typedef __attribute__((ext_vector_type(8)))  float    v8f;

#define EPSV 1e-5f

// ------------------------------- utilities --------------------------------

__global__ void fill_kernel(float* p, long long n, float v) {
  long long i = (long long)blockIdx.x * blockDim.x + threadIdx.x;
  if (i < n) p[i] = v;
}

__global__ void copy_kernel(const float* a, float* b, long long n) {
  long long i = (long long)blockIdx.x * blockDim.x + threadIdx.x;
  if (i < n) b[i] = a[i];
}

// ----------------------- per-column feature normalize ---------------------

__global__ void colstats_kernel(const float* __restrict__ x, int n, int D,
                                float* __restrict__ stats) {
  __shared__ float ss[64];
  if (threadIdx.x < 64) ss[threadIdx.x] = 0.0f;
  __syncthreads();
  long long total = (long long)n * D;
  for (long long i = (long long)blockIdx.x * blockDim.x + threadIdx.x; i < total;
       i += (long long)gridDim.x * blockDim.x) {
    float v = x[i];
    int c = (int)(i % D);
    atomicAdd(&ss[c], v);
    atomicAdd(&ss[32 + c], v * v);
  }
  __syncthreads();
  if (threadIdx.x < 64) atomicAdd(&stats[threadIdx.x], ss[threadIdx.x]);
}

__global__ void colnorm_kernel(float* __restrict__ x, int n, int D,
                               const float* __restrict__ stats) {
  long long i = (long long)blockIdx.x * blockDim.x + threadIdx.x;
  if (i >= (long long)n * D) return;
  int c = (int)(i % D);
  float mu = stats[c] / (float)n;
  float var = stats[32 + c] / (float)n - mu * mu;
  x[i] = (x[i] - mu) * rsqrtf(var + EPSV);
}

// --------------------------- edge-weight normalize ------------------------

__global__ void deg_kernel(const int* __restrict__ dst, const float* __restrict__ w,
                           float* __restrict__ deg, int E) {
  int e = blockIdx.x * blockDim.x + threadIdx.x;
  if (e < E) atomicAdd(&deg[dst[e]], w[e]);
}

__global__ void dinv_kernel(float* __restrict__ deg, int n) {
  int i = blockIdx.x * blockDim.x + threadIdx.x;
  if (i < n) {
    float d = deg[i];
    deg[i] = (d > 0.0f) ? rsqrtf(d) : 0.0f;
  }
}

__global__ void norm_kernel(const int* __restrict__ src, const int* __restrict__ dst,
                            const float* __restrict__ w, const float* __restrict__ dinv,
                            float* __restrict__ nrm, int E) {
  int e = blockIdx.x * blockDim.x + threadIdx.x;
  if (e < E) nrm[e] = dinv[src[e]] * w[e] * dinv[dst[e]];
}

// scatter hop: hout[dst] += norm[e] * hin[src]   (segment_sum semantics)
__global__ void spmm_kernel(const float* __restrict__ nrm, const int* __restrict__ src,
                            const int* __restrict__ dst, const float* __restrict__ hin,
                            float* __restrict__ hout, int E, int D) {
  long long t = (long long)blockIdx.x * blockDim.x + threadIdx.x;
  if (t >= (long long)E * D) return;
  int e = (int)(t / D);
  int f = (int)(t % D);
  atomicAdd(&hout[(long long)dst[e] * D + f], nrm[e] * hin[(long long)src[e] * D + f]);
}

// ------------------------------ WMMA GEMM ---------------------------------
// C[M,32] = act( A[M,32] @ B[32,32] + Cin + bias )
// One wave handles a 16-row tile and both 16-col tiles. Compile-time epilogue
// flags; wave-uniform full-tile fast path keeps the hot loop branch-free.
template <bool HAS_CIN, bool HAS_BIAS, bool RELU>
__global__ void gemm32_wmma_kernel(const float* __restrict__ A, const float* __restrict__ B,
                                   const float* __restrict__ Cin, const float* __restrict__ bias,
                                   float* __restrict__ Cout, int M) {
  int wave = blockIdx.x * (blockDim.x >> 5) + (threadIdx.x >> 5);
  int rowBase = wave * 16;
  if (rowBase >= M) return;
  const bool full = (rowBase + 16 <= M);   // wave-uniform
  int lane = threadIdx.x & 31;
  int half = lane >> 4;
  int lm   = lane & 15;

  int arow = rowBase + lm;
  if (!full && arow > M - 1) arow = M - 1;
  const float* ap = A + (size_t)arow * 32;

  // A 16x32 f16 layout: lanes 0-15 -> K 0-15, lanes 16-31 -> K 16-31
  v16h a;
  int k0 = half * 16;
  int k1 = half * 16 + 8;
#pragma unroll
  for (int j = 0; j < 8; ++j) {
    a[j]     = (_Float16)ap[k0 + j];
    a[8 + j] = (_Float16)ap[k1 + j];
  }

  int kb = half * 16;  // B: lanes 0-15 hold K 0-15, lanes 16-31 hold K 16-31
#pragma unroll
  for (int t = 0; t < 2; ++t) {
    int col = t * 16 + lm;
    v16h b;
#pragma unroll
    for (int j = 0; j < 16; ++j) b[j] = (_Float16)B[(kb + j) * 32 + col];

    v8f c = {};
    if (HAS_CIN) {
      if (full) {
#pragma unroll
        for (int r = 0; r < 8; ++r)
          c[r] = Cin[(size_t)(rowBase + half * 8 + r) * 32 + col];
      } else {
#pragma unroll
        for (int r = 0; r < 8; ++r) {
          int rr = rowBase + half * 8 + r;
          c[r] = (rr < M) ? Cin[(size_t)rr * 32 + col] : 0.0f;
        }
      }
    }
    c = __builtin_amdgcn_wmma_f32_16x16x32_f16(false, a, false, b, (short)0, c, false, false);

    float bv = HAS_BIAS ? bias[col] : 0.0f;
    if (full) {
#pragma unroll
      for (int r = 0; r < 8; ++r) {
        float v = c[r] + bv;
        if (RELU) v = fmaxf(v, 0.0f);
        Cout[(size_t)(rowBase + half * 8 + r) * 32 + col] = v;
      }
    } else {
#pragma unroll
      for (int r = 0; r < 8; ++r) {
        int rr = rowBase + half * 8 + r;
        if (rr < M) {
          float v = c[r] + bv;
          if (RELU) v = fmaxf(v, 0.0f);
          Cout[(size_t)rr * 32 + col] = v;
        }
      }
    }
  }
}

// K=1 "GEMM": out[n,32] = act(Cin + A[n]*W[0,32] + bias)
__global__ void gemm_k1_kernel(const float* __restrict__ A, const float* __restrict__ W,
                               const float* __restrict__ Cin, const float* __restrict__ bias,
                               float* __restrict__ Cout, int n, int relu) {
  long long t = (long long)blockIdx.x * blockDim.x + threadIdx.x;
  if (t >= (long long)n * 32) return;
  int r = (int)(t >> 5), j = (int)(t & 31);
  float v = (Cin ? Cin[t] : 0.0f) + A[r] * W[j];
  if (bias) v += bias[j];
  if (relu) v = fmaxf(v, 0.0f);
  Cout[t] = v;
}

// Dout=1 GEMM: out[n] = relu(A[n,32] @ w[32,1] + b)
__global__ void gemm_d1_kernel(const float* __restrict__ A, const float* __restrict__ w,
                               const float* __restrict__ b, float* __restrict__ out, int n) {
  int r = blockIdx.x * blockDim.x + threadIdx.x;
  if (r >= n) return;
  float acc = b[0];
  const float* ap = A + (size_t)r * 32;
#pragma unroll
  for (int k = 0; k < 32; ++k) acc += ap[k] * w[k];
  out[r] = fmaxf(acc, 0.0f);
}

// ------------- fused edge-MLP layer 1 (K = 32+32+1), WMMA ------------------
// eh[e,32] = relu([x[src] | x[dst] | ea] @ w1(65x32) + b1)
__global__ void edge_mlp1_wmma_kernel(const float* __restrict__ X, const int* __restrict__ src,
                                      const int* __restrict__ dst, const float* __restrict__ ea,
                                      const float* __restrict__ w1, const float* __restrict__ b1,
                                      float* __restrict__ eh, int E) {
  int wave = blockIdx.x * (blockDim.x >> 5) + (threadIdx.x >> 5);
  int rowBase = wave * 16;
  if (rowBase >= E) return;
  const bool full = (rowBase + 16 <= E);   // wave-uniform
  int lane = threadIdx.x & 31;
  int half = lane >> 4;
  int lm   = lane & 15;

  int e = rowBase + lm;
  if (!full && e > E - 1) e = E - 1;
  const float* xs = X + (size_t)src[e] * 32;
  const float* xd = X + (size_t)dst[e] * 32;

  v16h as, ad;
  int k0 = half * 16;
  int k1 = half * 16 + 8;
#pragma unroll
  for (int j = 0; j < 8; ++j) {
    as[j]     = (_Float16)xs[k0 + j];
    as[8 + j] = (_Float16)xs[k1 + j];
    ad[j]     = (_Float16)xd[k0 + j];
    ad[8 + j] = (_Float16)xd[k1 + j];
  }

  int kb = half * 16;
#pragma unroll
  for (int t = 0; t < 2; ++t) {
    int col = t * 16 + lm;
    v16h bs, bd;
#pragma unroll
    for (int j = 0; j < 16; ++j) {
      bs[j] = (_Float16)w1[(kb + j) * 32 + col];        // rows 0..31  (x[src] part)
      bd[j] = (_Float16)w1[(32 + kb + j) * 32 + col];   // rows 32..63 (x[dst] part)
    }
    float bcol = b1[col];
    float wea  = w1[64 * 32 + col];                     // row 64: edge_attr part
    v8f c;
#pragma unroll
    for (int r = 0; r < 8; ++r) {
      int er = rowBase + half * 8 + r;
      if (!full && er > E - 1) er = E - 1;
      c[r] = bcol + ea[er] * wea;
    }
    c = __builtin_amdgcn_wmma_f32_16x16x32_f16(false, as, false, bs, (short)0, c, false, false);
    c = __builtin_amdgcn_wmma_f32_16x16x32_f16(false, ad, false, bd, (short)0, c, false, false);
    if (full) {
#pragma unroll
      for (int r = 0; r < 8; ++r)
        eh[(size_t)(rowBase + half * 8 + r) * 32 + col] = fmaxf(c[r], 0.0f);
    } else {
#pragma unroll
      for (int r = 0; r < 8; ++r) {
        int er = rowBase + half * 8 + r;
        if (er < E) eh[(size_t)er * 32 + col] = fmaxf(c[r], 0.0f);
      }
    }
  }
}

// last layer's edge-MLP layer 1 (node dim = 1, K = 3)
__global__ void edge_mlp1_small_kernel(const float* __restrict__ X, const int* __restrict__ src,
                                       const int* __restrict__ dst, const float* __restrict__ ea,
                                       const float* __restrict__ w1, const float* __restrict__ b1,
                                       float* __restrict__ eh, int E) {
  int e = blockIdx.x * blockDim.x + threadIdx.x;
  if (e >= E) return;
  float sv = X[src[e]], dv = X[dst[e]], av = ea[e];
  float* p = eh + (size_t)e * 32;
#pragma unroll
  for (int j = 0; j < 32; ++j)
    p[j] = fmaxf(sv * w1[j] + dv * w1[32 + j] + av * w1[64 + j] + b1[j], 0.0f);
}

__global__ void layernorm_kernel(float* __restrict__ h, const float* __restrict__ g,
                                 const float* __restrict__ b, int E) {
  int e = blockIdx.x * blockDim.x + threadIdx.x;
  if (e >= E) return;
  float* p = h + (size_t)e * 32;
  float s = 0.0f, s2 = 0.0f;
#pragma unroll
  for (int j = 0; j < 32; ++j) {
    float v = p[j];
    s += v;
    s2 += v * v;
  }
  float mu = s * (1.0f / 32.0f);
  float var = s2 * (1.0f / 32.0f) - mu * mu;
  float rs = rsqrtf(var + EPSV);
#pragma unroll
  for (int j = 0; j < 32; ++j) p[j] = (p[j] - mu) * rs * g[j] + b[j];
}

__global__ void edge_w3_kernel(const float* __restrict__ h, const float* __restrict__ w3,
                               const float* __restrict__ b3, float* __restrict__ out, int E) {
  int e = blockIdx.x * blockDim.x + threadIdx.x;
  if (e >= E) return;
  float acc = b3[0];
  const float* p = h + (size_t)e * 32;
#pragma unroll
  for (int k = 0; k < 32; ++k) acc += p[k] * w3[k];
  out[e] = fmaxf(acc, 0.0f);
}

// ------------------------------ top-k --------------------------------------

__device__ __forceinline__ unsigned mapf(float f) {
  unsigned u = __float_as_uint(f);
  return (u & 0x80000000u) ? ~u : (u | 0x80000000u);
}

// single-block 32-step binary search on ordered float bits; no host readback
__global__ void topk_thresh_kernel(const float* __restrict__ xs, int n,
                                   const int* __restrict__ kptr, unsigned* __restrict__ out) {
  __shared__ unsigned s_lo, s_hi;
  __shared__ int s_cnt;
  int tid = threadIdx.x;
  int k = *kptr;
  if (tid == 0) { s_lo = 0u; s_hi = 0xFFFFFFFFu; }
  __syncthreads();
  for (int it = 0; it < 32; ++it) {
    if (tid == 0) s_cnt = 0;
    __syncthreads();
    unsigned mid = (unsigned)((((unsigned long long)s_lo + (unsigned long long)s_hi) + 1ull) >> 1);
    int c = 0;
    for (int i = tid; i < n; i += blockDim.x)
      if (mapf(xs[i]) >= mid) c++;
    atomicAdd(&s_cnt, c);
    __syncthreads();
    if (tid == 0) {
      if (s_cnt >= k) s_lo = mid; else s_hi = mid - 1u;
    }
    __syncthreads();
  }
  if (tid == 0) *out = s_lo;
}

__global__ void mark_kernel(const float* __restrict__ xs, const unsigned* __restrict__ tb,
                            float* __restrict__ out, int n) {
  int i = blockIdx.x * blockDim.x + threadIdx.x;
  if (i < n) out[i] = (mapf(xs[i]) >= *tb) ? 1.0f : 0.0f;
}

// ----------------------------- orchestration -------------------------------

extern "C" void kernel_launch(void* const* d_in, const int* in_sizes, int n_in,
                              void* d_out, int out_size, void* d_ws, size_t ws_size,
                              hipStream_t stream) {
  (void)n_in; (void)out_size; (void)ws_size;
  const int N = in_sizes[0];
  const int E = in_sizes[2];

  const float* x0  = (const float*)d_in[0];
  const int*   ei  = (const int*)d_in[1];
  const int*   src = ei;
  const int*   dst = ei + E;
  const float* ea0 = (const float*)d_in[2];
  const int*   kpt = (const int*)d_in[3];

  // pytree flatten order (sorted dict keys): edge[6]{b1,b2,b3,be1,be2,g1,g2,w1,w2,w3},
  // fc[6]{bs0..bs4, ws0..ws4}, tag[6]{W0..W3, b}
  const int PB = 4;
  auto EDG  = [&](int i, int f) { return (const float*)d_in[PB + i * 10 + f]; };
  auto FCP  = [&](int i, int f) { return (const float*)d_in[PB + 60 + i * 10 + f]; };
  auto TAGP = [&](int i, int f) { return (const float*)d_in[PB + 120 + i * 5 + f]; };

  float* ws = (float*)d_ws;
  size_t o = 0;
  float*    stats = ws + o; o += 64;
  unsigned* tbits = (unsigned*)(ws + o); o += 16;
  float* deg  = ws + o; o += (size_t)N;
  float* nrm  = ws + o; o += (size_t)E;
  float* eaA  = ws + o; o += (size_t)E;
  float* eaB  = ws + o; o += (size_t)E;
  float* acc  = ws + o; o += (size_t)N * 32;
  float* xA   = ws + o; o += (size_t)N * 32;
  float* xB   = ws + o; o += (size_t)N * 32;
  float* hA   = ws + o; o += (size_t)N * 32;
  float* hB   = ws + o; o += (size_t)N * 32;
  float* eh1  = ws + o; o += (size_t)E * 32;
  float* eh2  = ws + o; o += (size_t)E * 32;

  const int BS = 256;
  auto g1 = [&](long long n) { return dim3((unsigned)((n + BS - 1) / BS)); };
  int nwaves = (N + 15) / 16;
  dim3 ng((nwaves + 7) / 8);
  int ewaves = (E + 15) / 16;
  dim3 eg((ewaves + 7) / 8);

  copy_kernel<<<g1(N), BS, 0, stream>>>(x0, xA, N);
  copy_kernel<<<g1(E), BS, 0, stream>>>(ea0, eaA, E);

  float* xcur = xA;
  float* eacur = eaA;
  float* eanext = eaB;

  for (int i = 0; i < 6; ++i) {
    const int D = (i == 0) ? 1 : 32;

    // 1) per-column feature normalization (in place)
    fill_kernel<<<1, 64, 0, stream>>>(stats, 64, 0.0f);
    colstats_kernel<<<dim3(512), BS, 0, stream>>>(xcur, N, D, stats);
    colnorm_kernel<<<g1((long long)N * D), BS, 0, stream>>>(xcur, N, D, stats);

    // 2) symmetric edge normalization
    fill_kernel<<<g1(N), BS, 0, stream>>>(deg, N, 0.0f);
    deg_kernel<<<g1(E), BS, 0, stream>>>(dst, eacur, deg, E);
    dinv_kernel<<<g1(N), BS, 0, stream>>>(deg, N);
    norm_kernel<<<g1(E), BS, 0, stream>>>(src, dst, eacur, deg, nrm, E);

    // 3) TAGConv: out = sum_k h_k @ W_k ; x = relu(out + b)  (result -> xA)
    if (D == 1) {
      gemm_k1_kernel<<<g1((long long)N * 32), BS, 0, stream>>>(xcur, TAGP(i, 0), nullptr, nullptr, acc, N, 0);
      float* hin = xcur; float* ha = hA; float* hb = hB;
      for (int kk = 1; kk <= 3; ++kk) {
        fill_kernel<<<g1(N), BS, 0, stream>>>(ha, N, 0.0f);
        spmm_kernel<<<g1((long long)E), BS, 0, stream>>>(nrm, src, dst, hin, ha, E, 1);
        if (kk < 3)
          gemm_k1_kernel<<<g1((long long)N * 32), BS, 0, stream>>>(ha, TAGP(i, kk), acc, nullptr, acc, N, 0);
        else
          gemm_k1_kernel<<<g1((long long)N * 32), BS, 0, stream>>>(ha, TAGP(i, kk), acc, TAGP(i, 4), xA, N, 1);
        hin = ha; float* t2 = ha; ha = hb; hb = t2;
      }
    } else {
      gemm32_wmma_kernel<false, false, false><<<ng, BS, 0, stream>>>(xcur, TAGP(i, 0), nullptr, nullptr, acc, N);
      float* hin = xcur; float* ha = hA; float* hb = hB;
      for (int kk = 1; kk <= 3; ++kk) {
        fill_kernel<<<g1((long long)N * 32), BS, 0, stream>>>(ha, (long long)N * 32, 0.0f);
        spmm_kernel<<<g1((long long)E * 32), BS, 0, stream>>>(nrm, src, dst, hin, ha, E, 32);
        if (kk < 3)
          gemm32_wmma_kernel<true, false, false><<<ng, BS, 0, stream>>>(ha, TAGP(i, kk), acc, nullptr, acc, N);
        else
          gemm32_wmma_kernel<true, true, true><<<ng, BS, 0, stream>>>(ha, TAGP(i, kk), acc, TAGP(i, 4), xA, N);
        hin = ha; float* t2 = ha; ha = hb; hb = t2;
      }
    }

    // 4) FC stack (xA -> ... -> xB; last conv ends 32->1 into xB[0:N])
    if (i < 5) {
      float* fin = xA; float* fout = xB;
      for (int l = 0; l < 5; ++l) {
        gemm32_wmma_kernel<false, true, true><<<ng, BS, 0, stream>>>(fin, FCP(i, 5 + l), nullptr, FCP(i, l), fout, N);
        float* t2 = fin; fin = fout; fout = t2;
      }
    } else {
      float* fin = xA; float* fout = xB;
      for (int l = 0; l < 4; ++l) {
        gemm32_wmma_kernel<false, true, true><<<ng, BS, 0, stream>>>(fin, FCP(i, 5 + l), nullptr, FCP(i, l), fout, N);
        float* t2 = fin; fin = fout; fout = t2;
      }
      gemm_d1_kernel<<<g1(N), BS, 0, stream>>>(fin, FCP(i, 9), FCP(i, 4), xB, N);
    }

    // 5) edge MLP: eacur -> eanext
    if (i < 5)
      edge_mlp1_wmma_kernel<<<eg, BS, 0, stream>>>(xB, src, dst, eacur, EDG(i, 7), EDG(i, 0), eh1, E);
    else
      edge_mlp1_small_kernel<<<g1(E), BS, 0, stream>>>(xB, src, dst, eacur, EDG(i, 7), EDG(i, 0), eh1, E);
    layernorm_kernel<<<g1(E), BS, 0, stream>>>(eh1, EDG(i, 5), EDG(i, 3), E);
    gemm32_wmma_kernel<false, true, true><<<eg, BS, 0, stream>>>(eh1, EDG(i, 8), nullptr, EDG(i, 1), eh2, E);
    layernorm_kernel<<<g1(E), BS, 0, stream>>>(eh2, EDG(i, 6), EDG(i, 4), E);
    edge_w3_kernel<<<g1(E), BS, 0, stream>>>(eh2, EDG(i, 9), EDG(i, 2), eanext, E);
    { float* t2 = eacur; eacur = eanext; eanext = t2; }

    xcur = xB;
  }

  // 6) outputs: top-k indicator over final node scores (xB[0:N]), then edge_attr
  float* outv = (float*)d_out;
  topk_thresh_kernel<<<1, 1024, 0, stream>>>(xB, N, kpt, tbits);
  mark_kernel<<<g1(N), BS, 0, stream>>>(xB, tbits, outv, N);
  copy_kernel<<<g1(E), BS, 0, stream>>>(eacur, outv + N, E);
}
